// Non_local_81320910782920
// MI455X (gfx1250) — matrix operand order repaired and checked by reading, not compiled, not run on hardware
//
#include <hip/hip_runtime.h>
#include <hip/hip_bf16.h>

// ---------------------------------------------------------------------------
// Non-local block (linear attention) for MI455X / gfx1250, bf16 WMMA pipeline.
// y = theta @ (phi @ g) / N   (associativity removes the NxN matrix)
// Double-buffered LDS, one barrier per K-step, async global->LDS for raw bf16
// copies, hardware f32->bf16 converts, v_wmma_f32_16x16x32_bf16 compute.
// ---------------------------------------------------------------------------

typedef __attribute__((ext_vector_type(16))) __bf16 v16bf;
typedef __attribute__((ext_vector_type(8)))  float  v8f;

#define B_   16
#define C_   256
#define CI_  128
#define N_   4096
#define LDA_ 40   // padded LDS stride (halfwords) -> conflict-free b128 frag reads

__device__ __forceinline__ unsigned short f2bf(float f) {
    __bf16 h = (__bf16)f;                      // hardware fptrunc (RNE)
    return __builtin_bit_cast(unsigned short, h);
}
__device__ __forceinline__ unsigned pk2bf(float lo, float hi) {
    return (unsigned)f2bf(lo) | ((unsigned)f2bf(hi) << 16);
}

__device__ __forceinline__ void async_wait() {
    asm volatile("s_wait_asynccnt 0x0" ::: "memory");
}

struct U32x2 { uint4 a, b; };                  // 32-byte carrier for a fragment

__device__ __forceinline__ v16bf ldfragA(const unsigned short* As, int row, int kh) {
    U32x2 u;
    u.a = *(const uint4*)(As + row * LDA_ + kh * 8);        // K = kh*8 .. +7
    u.b = *(const uint4*)(As + row * LDA_ + 16 + kh * 8);   // K = 16+kh*8 .. +7
    return __builtin_bit_cast(v16bf, u);
}
__device__ __forceinline__ v16bf ldfragB(const unsigned short* Bs, int col, int kh) {
    U32x2 u;
    u.a = *(const uint4*)(Bs + col * LDA_ + kh * 8);        // K = kh*8 .. +7
    u.b = *(const uint4*)(Bs + col * LDA_ + 16 + kh * 8);   // K = 16+kh*8 .. +7
    return __builtin_bit_cast(v16bf, u);
}

// ---- staging helpers (256 threads, tile = 128 rows x 32 K halfwords) -------
// bf16 source, K contiguous: async DMA global -> LDS (no VGPR round-trip)
__device__ __forceinline__ void stage_bf16_async(unsigned short* dst,
                                                 const unsigned short* src, int ld, int kc) {
    int t = threadIdx.x;
#pragma unroll
    for (int i = 0; i < 2; ++i) {
        int idx = t + i * 256;                 // 512 x 16B chunks
        int row = idx >> 2, kq = idx & 3;
        const unsigned short* g = src + row * ld + kc + kq * 8;
        unsigned lds = (unsigned)(uintptr_t)(dst + row * LDA_ + kq * 8);
        asm volatile("global_load_async_to_lds_b128 %0, %1, off"
                     :: "v"(lds), "v"(g) : "memory");
    }
}
// f32 source, K contiguous: convert + copy
__device__ __forceinline__ void stage_f32(unsigned short* dst,
                                          const float* src, int ld, int kc) {
    int t = threadIdx.x;
#pragma unroll
    for (int i = 0; i < 4; ++i) {
        int idx = t + i * 256;                 // 1024 x float4
        int row = idx >> 3, kq = idx & 7;
        float4 v = *(const float4*)(src + row * ld + kc + kq * 4);
        uint2 pk;
        pk.x = pk2bf(v.x, v.y);
        pk.y = pk2bf(v.z, v.w);
        *(uint2*)(dst + row * LDA_ + kq * 4) = pk;
    }
}
// bf16 source laid out [K][col] (col contiguous): transpose into Bs[col][K]
__device__ __forceinline__ void stageT_bf16(unsigned short* dst,
                                            const unsigned short* src, int ld, int kbase) {
    int t = threadIdx.x;
#pragma unroll
    for (int i = 0; i < 2; ++i) {
        int idx = t + i * 256;                 // 32 K x 16 col-groups
        int k = idx >> 4, cg = idx & 15;
        union { uint4 q; unsigned short e[8]; } u;
        u.q = *(const uint4*)(src + (kbase + k) * ld + cg * 8);
#pragma unroll
        for (int j = 0; j < 8; ++j) dst[(cg * 8 + j) * LDA_ + k] = u.e[j];
    }
}
// f32 source laid out [K][col] (col contiguous): convert + transpose
__device__ __forceinline__ void stageT_f32(unsigned short* dst,
                                           const float* src, int ld, int kbase) {
    int t = threadIdx.x;
#pragma unroll
    for (int i = 0; i < 4; ++i) {
        int idx = t + i * 256;                 // 32 K x 32 col-groups
        int k = idx >> 5, cg = idx & 31;
        float4 v = *(const float4*)(src + (kbase + k) * ld + cg * 4);
        dst[(cg * 4 + 0) * LDA_ + k] = f2bf(v.x);
        dst[(cg * 4 + 1) * LDA_ + k] = f2bf(v.y);
        dst[(cg * 4 + 2) * LDA_ + k] = f2bf(v.z);
        dst[(cg * 4 + 3) * LDA_ + k] = f2bf(v.w);
    }
}

// ---- WMMA micro-core: 8 waves, wave grid 2x4, wave tile 64x32 --------------
__device__ __forceinline__ void mma_step(const unsigned short* As,
                                         const unsigned short* Bs, v8f acc[4][2]) {
    int lane = threadIdx.x & 31;
    int wave = threadIdx.x >> 5;
    int wr = wave >> 2, wc = wave & 3;
    int lrow = lane & 15, kh = lane >> 4;
    v16bf a[4], b[2];
#pragma unroll
    for (int t = 0; t < 4; ++t) a[t] = ldfragA(As, wr * 64 + t * 16 + lrow, kh);
#pragma unroll
    for (int u = 0; u < 2; ++u) b[u] = ldfragB(Bs, wc * 32 + u * 16 + lrow, kh);
#pragma unroll
    for (int t = 0; t < 4; ++t)
#pragma unroll
        for (int u = 0; u < 2; ++u)
            acc[t][u] = __builtin_amdgcn_wmma_f32_16x16x32_bf16(
                false, a[t], false, b[u], (short)0, acc[t][u], false, false);
}

__device__ __forceinline__ void acc_zero(v8f acc[4][2]) {
#pragma unroll
    for (int t = 0; t < 4; ++t)
#pragma unroll
        for (int u = 0; u < 2; ++u)
#pragma unroll
            for (int e = 0; e < 8; ++e) acc[t][u][e] = 0.f;
}

// ---------------------------------------------------------------------------
// K1: projections  {g,theta,phi} = W(128x256) @ x(256xN) + b
// ---------------------------------------------------------------------------
__global__ __launch_bounds__(256) void proj_kernel(
    const float* __restrict__ x,
    const float* __restrict__ g_w, const float* __restrict__ g_b,
    const float* __restrict__ th_w, const float* __restrict__ th_b,
    const float* __restrict__ ph_w, const float* __restrict__ ph_b,
    unsigned short* __restrict__ g_t, unsigned short* __restrict__ th_t,
    unsigned short* __restrict__ phi) {
    __shared__ __align__(16) unsigned short As[2][128 * LDA_];
    __shared__ __align__(16) unsigned short Bs[2][128 * LDA_];
    int nBase = blockIdx.x * 128;
    int b = blockIdx.y;
    int p = blockIdx.z;
    const float* wsel = (p == 0) ? g_w : (p == 1) ? th_w : ph_w;
    const float* bsel = (p == 0) ? g_b : (p == 1) ? th_b : ph_b;
    const float* xb = x + (size_t)b * C_ * N_ + nBase;

    v8f acc[4][2];
    acc_zero(acc);
    stage_f32(As[0], wsel, C_, 0);
    stageT_f32(Bs[0], xb, N_, 0);
    __syncthreads();
    const int nIter = C_ / 32;
#pragma unroll
    for (int it = 0; it < nIter; ++it) {
        const int cur = it & 1;
        if (it + 1 < nIter) {
            stage_f32(As[cur ^ 1], wsel, C_, (it + 1) * 32);
            stageT_f32(Bs[cur ^ 1], xb, N_, (it + 1) * 32);
        }
        mma_step(As[cur], Bs[cur], acc);
        __syncthreads();
    }
    int lane = threadIdx.x & 31, wave = threadIdx.x >> 5;
    int wr = wave >> 2, wc = wave & 3, lrow = lane & 15, kh = lane >> 4;
    // branchless store path: dst/strides resolved once per block
    unsigned short* dst;
    int rs, cs;
    size_t boff;
    if (p == 2) { dst = phi + (size_t)b * CI_ * N_; rs = N_; cs = 1; boff = nBase; }
    else {
        dst = ((p == 0) ? g_t : th_t) + (size_t)b * N_ * CI_;
        rs = 1; cs = CI_; boff = (size_t)nBase * CI_;
    }
#pragma unroll
    for (int t = 0; t < 4; ++t)
#pragma unroll
        for (int u = 0; u < 2; ++u)
#pragma unroll
            for (int r = 0; r < 8; ++r) {
                int row = wr * 64 + t * 16 + kh * 8 + r;   // output channel
                int col = wc * 32 + u * 16 + lrow;         // n (local)
                dst[boff + (size_t)row * rs + (size_t)col * cs] =
                    f2bf(acc[t][u][r] + bsel[row]);
            }
}

// ---------------------------------------------------------------------------
// K2: split-K partials of  M = phi(128xN) @ g(NxCi)
// ---------------------------------------------------------------------------
__global__ __launch_bounds__(256) void mpart_kernel(
    const unsigned short* __restrict__ phi, const unsigned short* __restrict__ g_t,
    float* __restrict__ Mpart) {
    __shared__ __align__(16) unsigned short As[2][128 * LDA_];
    __shared__ __align__(16) unsigned short Bs[2][128 * LDA_];
    int split = blockIdx.x, b = blockIdx.y;
    const unsigned short* pb = phi + (size_t)b * CI_ * N_;
    const unsigned short* gb = g_t + (size_t)b * N_ * CI_;
    int kc0 = split * 256;

    v8f acc[4][2];
    acc_zero(acc);
    stage_bf16_async(As[0], pb, N_, kc0);      // A: phi [Ci][N], K=n contiguous
    stageT_bf16(Bs[0], gb, CI_, kc0);          // B: g_t [n][cj] -> [cj][k]
    async_wait();
    __syncthreads();
#pragma unroll
    for (int it = 0; it < 8; ++it) {
        const int cur = it & 1;
        if (it + 1 < 8) {
            stage_bf16_async(As[cur ^ 1], pb, N_, kc0 + (it + 1) * 32);
            stageT_bf16(Bs[cur ^ 1], gb, CI_, kc0 + (it + 1) * 32);
        }
        mma_step(As[cur], Bs[cur], acc);
        async_wait();
        __syncthreads();
    }
    int lane = threadIdx.x & 31, wave = threadIdx.x >> 5;
    int wr = wave >> 2, wc = wave & 3, lrow = lane & 15, kh = lane >> 4;
    float* dst = Mpart + ((size_t)split * B_ + b) * (CI_ * CI_);
#pragma unroll
    for (int t = 0; t < 4; ++t)
#pragma unroll
        for (int u = 0; u < 2; ++u)
#pragma unroll
            for (int r = 0; r < 8; ++r) {
                int row = wr * 64 + t * 16 + kh * 8 + r;
                int col = wc * 32 + u * 16 + lrow;
                dst[row * CI_ + col] = acc[t][u][r];
            }
}

// K2b: reduce split-K partials, fold 1/N, emit bf16 M
__global__ __launch_bounds__(256) void mreduce_kernel(
    const float* __restrict__ Mpart, unsigned short* __restrict__ Mbf) {
    int i = blockIdx.x * 256 + threadIdx.x;      // 16 * 16384 total
    int b = i >> 14, idx = i & 16383;
    float s = 0.f;
#pragma unroll
    for (int p = 0; p < 16; ++p) s += Mpart[((size_t)p * B_ + b) * (CI_ * CI_) + idx];
    Mbf[(size_t)b * (CI_ * CI_) + idx] = f2bf(s * (1.0f / N_));
}

// ---------------------------------------------------------------------------
// K3: y = theta(NxCi) @ M(CixCi), stored [N][Ci] bf16
// ---------------------------------------------------------------------------
__global__ __launch_bounds__(256) void ygemm_kernel(
    const unsigned short* __restrict__ th_t, const unsigned short* __restrict__ Mbf,
    unsigned short* __restrict__ ynat) {
    __shared__ __align__(16) unsigned short As[2][128 * LDA_];
    __shared__ __align__(16) unsigned short Bs[2][128 * LDA_];
    int nBase = blockIdx.x * 128, b = blockIdx.y;
    const unsigned short* thb = th_t + (size_t)b * N_ * CI_ + (size_t)nBase * CI_;
    const unsigned short* mb = Mbf + (size_t)b * (CI_ * CI_);

    v8f acc[4][2];
    acc_zero(acc);
    stage_bf16_async(As[0], thb, CI_, 0);      // A: theta rows n, K=ci contiguous
    stageT_bf16(Bs[0], mb, CI_, 0);            // B: M [ci][cj] -> [cj][k]
    async_wait();
    __syncthreads();
#pragma unroll
    for (int it = 0; it < 4; ++it) {
        const int cur = it & 1;
        if (it + 1 < 4) {
            stage_bf16_async(As[cur ^ 1], thb, CI_, (it + 1) * 32);
            stageT_bf16(Bs[cur ^ 1], mb, CI_, (it + 1) * 32);
        }
        mma_step(As[cur], Bs[cur], acc);
        async_wait();
        __syncthreads();
    }
    int lane = threadIdx.x & 31, wave = threadIdx.x >> 5;
    int wr = wave >> 2, wc = wave & 3, lrow = lane & 15, kh = lane >> 4;
    unsigned short* yb = ynat + (size_t)b * N_ * CI_;
#pragma unroll
    for (int t = 0; t < 4; ++t)
#pragma unroll
        for (int u = 0; u < 2; ++u)
#pragma unroll
            for (int r = 0; r < 8; ++r) {
                int row = wr * 64 + t * 16 + kh * 8 + r;   // n (local)
                int col = wc * 32 + u * 16 + lrow;         // cj
                yb[(size_t)(nBase + row) * CI_ + col] = f2bf(acc[t][u][r]);
            }
}

// ---------------------------------------------------------------------------
// K4: out = BN(w_w(256x128) @ y(128xN) + w_b) + x
// ---------------------------------------------------------------------------
__global__ __launch_bounds__(256) void outgemm_kernel(
    const float* __restrict__ w_w, const unsigned short* __restrict__ ynat,
    const float* __restrict__ w_b,
    const float* __restrict__ gmm, const float* __restrict__ bta,
    const float* __restrict__ mu, const float* __restrict__ var,
    const float* __restrict__ x, float* __restrict__ out) {
    __shared__ __align__(16) unsigned short As[2][128 * LDA_];
    __shared__ __align__(16) unsigned short Bs[2][128 * LDA_];
    int nBase = blockIdx.x * 128;
    int cBase = blockIdx.y * 128;
    int b = blockIdx.z;
    const float* wwb = w_w + (size_t)cBase * CI_;
    const unsigned short* yb = ynat + (size_t)b * N_ * CI_ + (size_t)nBase * CI_;

    v8f acc[4][2];
    acc_zero(acc);
    stage_f32(As[0], wwb, CI_, 0);             // A: w_w rows c, K=ci
    stage_bf16_async(Bs[0], yb, CI_, 0);       // B: y [n][ci] direct async copy
    async_wait();
    __syncthreads();
#pragma unroll
    for (int it = 0; it < 4; ++it) {
        const int cur = it & 1;
        if (it + 1 < 4) {
            stage_f32(As[cur ^ 1], wwb, CI_, (it + 1) * 32);
            stage_bf16_async(Bs[cur ^ 1], yb, CI_, (it + 1) * 32);
        }
        mma_step(As[cur], Bs[cur], acc);
        async_wait();
        __syncthreads();
    }
    int lane = threadIdx.x & 31, wave = threadIdx.x >> 5;
    int wr = wave >> 2, wc = wave & 3, lrow = lane & 15, kh = lane >> 4;
    const float* xb = x + (size_t)b * C_ * N_;
    float* ob = out + (size_t)b * C_ * N_;
#pragma unroll
    for (int t = 0; t < 4; ++t)
#pragma unroll
        for (int u = 0; u < 2; ++u)
#pragma unroll
            for (int r = 0; r < 8; ++r) {
                int c = cBase + wr * 64 + t * 16 + kh * 8 + r;
                int n = nBase + wc * 32 + u * 16 + lrow;
                float inv = gmm[c] * rsqrtf(var[c] + 1e-5f);
                float add = bta[c] - mu[c] * inv;
                float v = (acc[t][u][r] + w_b[c]) * inv + add + xb[(size_t)c * N_ + n];
                ob[(size_t)c * N_ + n] = v;
            }
}

// ---------------------------------------------------------------------------
extern "C" void kernel_launch(void* const* d_in, const int* in_sizes, int n_in,
                              void* d_out, int out_size, void* d_ws, size_t ws_size,
                              hipStream_t stream) {
    const float* x    = (const float*)d_in[0];
    const float* g_w  = (const float*)d_in[1];
    const float* g_b  = (const float*)d_in[2];
    const float* th_w = (const float*)d_in[3];
    const float* th_b = (const float*)d_in[4];
    const float* ph_w = (const float*)d_in[5];
    const float* ph_b = (const float*)d_in[6];
    const float* w_w  = (const float*)d_in[7];
    const float* w_b  = (const float*)d_in[8];
    const float* gmm  = (const float*)d_in[9];
    const float* bta  = (const float*)d_in[10];
    const float* mu   = (const float*)d_in[11];
    const float* var  = (const float*)d_in[12];
    float* out = (float*)d_out;

    const size_t SZP = (size_t)B_ * N_ * CI_;       // halfword count per bf16 buffer
    unsigned short* g_t  = (unsigned short*)d_ws;
    unsigned short* th_t = g_t + SZP;
    unsigned short* phi  = th_t + SZP;
    unsigned short* ynat = phi + SZP;
    float* Mpart = (float*)(ynat + SZP);            // [16 split][B][128][128] f32
    unsigned short* Mbf = (unsigned short*)(Mpart + (size_t)16 * B_ * CI_ * CI_);

    proj_kernel<<<dim3(N_ / 128, B_, 3), 256, 0, stream>>>(
        x, g_w, g_b, th_w, th_b, ph_w, ph_b, g_t, th_t, phi);
    mpart_kernel<<<dim3(16, B_), 256, 0, stream>>>(phi, g_t, Mpart);
    mreduce_kernel<<<dim3((B_ * CI_ * CI_) / 256), 256, 0, stream>>>(Mpart, Mbf);
    ygemm_kernel<<<dim3(N_ / 128, B_), 256, 0, stream>>>(th_t, Mbf, ynat);
    outgemm_kernel<<<dim3(N_ / 128, C_ / 128, B_), 256, 0, stream>>>(
        w_w, ynat, w_b, gmm, bta, mu, var, x, out);
}
